// SpatialAttention3D_74483322847420
// MI455X (gfx1250) — matrix operand order repaired and checked by reading, not compiled, not run on hardware
//
#include <hip/hip_runtime.h>

#define CCH 128
#define GRP 8
#define NTOK 8192
#define BB 2
#define EPSV 1e-5f

typedef __attribute__((ext_vector_type(16))) __bf16 v16bf;
typedef __attribute__((ext_vector_type(8)))  __bf16 v8bf;
typedef __attribute__((ext_vector_type(8)))  float  v8f;

static __device__ inline v8f vzero() {
  v8f z = {0.f,0.f,0.f,0.f,0.f,0.f,0.f,0.f};
  return z;
}

static __device__ inline v8f wmma_bf16(v16bf a, v16bf b, v8f c) {
  // D = A(16x32) * B(32x16) + C, f32 accumulate
  return __builtin_amdgcn_wmma_f32_16x16x32_bf16(false, a, false, b, (short)0, c, false, false);
}

static __device__ inline v16bf cat8(v8bf lo, v8bf hi) {
  v16bf r;
#pragma unroll
  for (int i = 0; i < 8; ++i) { r[i] = lo[i]; r[i + 8] = hi[i]; }
  return r;
}

// A-fragment (16x32 bf16, M=rows) from row-major bf16 src with leading dim ld.
// Lane l: row = l&15; k-slices {kb..kb+7, kb+16..kb+23}, kb = 8*(l>>4).
static __device__ inline v16bf load_a_bf16(const __bf16* src, int ld, int lane) {
  const int row = lane & 15;
  const int kb  = (lane >> 4) * 8;
  const __bf16* p = src + (size_t)row * ld + kb;
  v8bf lo = *(const v8bf*)p;
  v8bf hi = *(const v8bf*)(p + 16);
  return cat8(lo, hi);
}

// B-fragment (32x16 bf16) loaded from row-major B^T source (src[n][k], ld elems).
// Lane l: n = l&15; k-slices {kb..kb+7, kb+16..kb+23}, kb = 8*(l>>4).
static __device__ inline v16bf load_bt_bf16(const __bf16* src, int ld, int lane) {
  const int n  = lane & 15;
  const int kb = (lane >> 4) * 8;
  const __bf16* p = src + (size_t)n * ld + kb;
  v8bf lo = *(const v8bf*)p;
  v8bf hi = *(const v8bf*)(p + 16);
  return cat8(lo, hi);
}

// Same B^T load but from f32 source with convert (weights).
static __device__ inline v16bf load_bt_f32(const float* src, int ld, int lane) {
  const int n  = lane & 15;
  const int kb = (lane >> 4) * 8;
  const float* p = src + (size_t)n * ld + kb;
  v16bf r;
#pragma unroll
  for (int i = 0; i < 16; ++i) r[i] = (__bf16)p[(i & 7) + ((i >> 3) << 4)];
  return r;
}

// CDNA5 async tensor path: 16B global -> LDS without a VGPR round-trip.
// GV addressing: 64-bit address in a VGPR pair, LDS byte offset in VDST.
static __device__ inline void async_load16(unsigned lds_off, const void* gaddr) {
  asm volatile("global_load_async_to_lds_b128 %0, %1, off"
               :
               : "v"(lds_off), "v"((unsigned long long)(uintptr_t)gaddr)
               : "memory");
}

// ---------------- Kernel 1: GroupNorm statistics ----------------
__global__ __launch_bounds__(256) void gn_stats_k(const float* __restrict__ x,
                                                  float* __restrict__ stats) {
  __shared__ float ssum[256];
  __shared__ float ssq[256];
  const int tid = threadIdx.x;
  // group channels are contiguous: block (b*8+g) reduces 16*NTOK contiguous floats
  const float4* base = (const float4*)(x + (size_t)blockIdx.x * 16 * NTOK);
  float s = 0.f, q = 0.f;
  for (int i = tid; i < (16 * NTOK) / 4; i += 256) {
    float4 v = base[i];
    s += v.x + v.y + v.z + v.w;
    q += v.x * v.x + v.y * v.y + v.z * v.z + v.w * v.w;
  }
  ssum[tid] = s; ssq[tid] = q;
  __syncthreads();
  for (int st = 128; st > 0; st >>= 1) {
    if (tid < st) { ssum[tid] += ssum[tid + st]; ssq[tid] += ssq[tid + st]; }
    __syncthreads();
  }
  if (tid == 0) {
    const float n = 16.f * NTOK;
    const float mean = ssum[0] / n;
    const float var  = ssq[0] / n - mean * mean;
    stats[blockIdx.x * 2]     = mean;
    stats[blockIdx.x * 2 + 1] = rsqrtf(var + EPSV);
  }
}

// ---------------- Kernel 2: fused GroupNorm + QKV GEMM ----------------
__global__ __launch_bounds__(256) void qkv_k(const float* __restrict__ x,
                                             const float* __restrict__ gamma,
                                             const float* __restrict__ beta,
                                             const float* __restrict__ wqkv,
                                             const float* __restrict__ bqkv,
                                             const float* __restrict__ stats,
                                             __bf16* __restrict__ qo,
                                             __bf16* __restrict__ ko,
                                             __bf16* __restrict__ vto) {
  const int lane = threadIdx.x & 31;
  const int widx = threadIdx.x >> 5;
  const int wg   = blockIdx.x * 8 + widx;
  const int b    = wg / (NTOK / 16);
  const int tb   = (wg % (NTOK / 16)) * 16;
  const float* xb = x + (size_t)b * CCH * NTOK;

  float mr[GRP], rr[GRP];
#pragma unroll
  for (int g = 0; g < GRP; ++g) {
    mr[g] = stats[(b * GRP + g) * 2];
    rr[g] = stats[(b * GRP + g) * 2 + 1];
  }

  const int row  = lane & 15;
  const int half = lane >> 4;
  const int col  = lane & 15;
  const int token = tb + row;

  // A fragments: normalized activations, 16 tokens x 128 channels
  v16bf a[4];
#pragma unroll
  for (int kt = 0; kt < 4; ++kt) {
    const int kb = half * 8;
#pragma unroll
    for (int i = 0; i < 16; ++i) {
      const int c = kt * 32 + kb + (i & 7) + ((i >> 3) << 4);
      const float v = xb[(size_t)c * NTOK + token];
      const float hn = (v - mr[c >> 4]) * rr[c >> 4] * gamma[c] + beta[c];
      a[kt][i] = (__bf16)hn;
    }
  }

  __bf16* qb_ = qo  + (size_t)b * NTOK * CCH;   // [n][c]
  __bf16* kb_ = ko  + (size_t)b * NTOK * CCH;   // [n][c]
  __bf16* vb_ = vto + (size_t)b * CCH * NTOK;   // [c][n] (transposed)

  for (int ot = 0; ot < 24; ++ot) {
    v8f acc = vzero();
#pragma unroll
    for (int kt = 0; kt < 4; ++kt) {
      v16bf bt = load_bt_f32(wqkv + (size_t)(ot * 16) * CCH + kt * 32, CCH, lane);
      acc = wmma_bf16(a[kt], bt, acc);
    }
    const int o = ot * 16 + col;
    const float bias = bqkv[o];
    if (ot < 16) {
      __bf16* dst = (ot < 8) ? qb_ : kb_;
      const int oc = o & 127;
#pragma unroll
      for (int r = 0; r < 8; ++r) {
        const int t = tb + r + half * 8;
        dst[(size_t)t * CCH + oc] = (__bf16)(acc[r] + bias);
      }
    } else {
      const int c = o - 256;
#pragma unroll
      for (int r = 0; r < 8; ++r) {
        const int t = tb + r + half * 8;
        vb_[(size_t)c * NTOK + t] = (__bf16)(acc[r] + bias);
      }
    }
  }
}

// ---------------- Kernel 3: flash attention (async double-buffered) ----------------
__global__ __launch_bounds__(256) void attn_k(const __bf16* __restrict__ qi,
                                              const __bf16* __restrict__ ki,
                                              const __bf16* __restrict__ vti,
                                              __bf16* __restrict__ ho) {
  __shared__ __align__(16) __bf16 Kt[2][32 * 136];    // 32 keys x 128d (padded)
  __shared__ __align__(16) __bf16 Vt[2][128 * 40];    // 128c x 32 keys (padded)
  __shared__ __align__(16) __bf16 Pb[8 * 16 * 40];    // per-wave P transpose buffer

  const int tid  = threadIdx.x;
  const int lane = tid & 31;
  const int widx = tid >> 5;
  const int b    = blockIdx.x >> 6;
  const int q0   = (blockIdx.x & 63) * 128 + widx * 16;
  const int half = lane >> 4;
  const int col  = lane & 15;

  const __bf16* qb_ = qi  + (size_t)b * NTOK * CCH;
  const __bf16* kb_ = ki  + (size_t)b * NTOK * CCH;
  const __bf16* vb_ = vti + (size_t)b * CCH * NTOK;

  // per-thread static addressing for the cooperative async stage
  const int kr = tid >> 4, kch = tid & 15;        // K: rows 0..15 (j=0) / 16..31 (j=1)
  const int vc = tid >> 2, vpt = tid & 3;         // V: channels 0..63 / 64..127

  // Q fragments stay in registers for the whole pass
  v16bf aq[4];
#pragma unroll
  for (int kt = 0; kt < 4; ++kt)
    aq[kt] = load_a_bf16(qb_ + (size_t)q0 * CCH + kt * 32, CCH, lane);

  v8f O[8];
  float M[8], L[8];
#pragma unroll
  for (int t = 0; t < 8; ++t) O[t] = vzero();
#pragma unroll
  for (int r = 0; r < 8; ++r) { M[r] = -3.0e38f; L[r] = 0.f; }

  const float scale = 0.08838834764831845f;  // 1/sqrt(128)
  __bf16* Pw = Pb + widx * 16 * 40;
  const int NT = NTOK / 32;

  // issue tile 0 into buffer 0 (4 async-to-LDS instructions per wave)
  {
    async_load16((unsigned)(uintptr_t)&Kt[0][kr * 136 + kch * 8],
                 kb_ + (size_t)kr * CCH + kch * 8);
    async_load16((unsigned)(uintptr_t)&Kt[0][(kr + 16) * 136 + kch * 8],
                 kb_ + (size_t)(kr + 16) * CCH + kch * 8);
    async_load16((unsigned)(uintptr_t)&Vt[0][vc * 40 + vpt * 8],
                 vb_ + (size_t)vc * NTOK + vpt * 8);
    async_load16((unsigned)(uintptr_t)&Vt[0][(vc + 64) * 40 + vpt * 8],
                 vb_ + (size_t)(vc + 64) * NTOK + vpt * 8);
  }

  for (int kt = 0; kt < NT; ++kt) {
    const int bt = kt & 1;
    __syncthreads();  // everyone is done reading buffer bt^1 (previous tile)
    if (kt + 1 < NT) {
      const size_t nb = (size_t)(kt + 1) * 32;
      async_load16((unsigned)(uintptr_t)&Kt[bt ^ 1][kr * 136 + kch * 8],
                   kb_ + (nb + kr) * CCH + kch * 8);
      async_load16((unsigned)(uintptr_t)&Kt[bt ^ 1][(kr + 16) * 136 + kch * 8],
                   kb_ + (nb + kr + 16) * CCH + kch * 8);
      async_load16((unsigned)(uintptr_t)&Vt[bt ^ 1][vc * 40 + vpt * 8],
                   vb_ + (size_t)vc * NTOK + nb + vpt * 8);
      async_load16((unsigned)(uintptr_t)&Vt[bt ^ 1][(vc + 64) * 40 + vpt * 8],
                   vb_ + (size_t)(vc + 64) * NTOK + nb + vpt * 8);
      // in-order completion: <=4 outstanding means this tile's 4 loads landed
      asm volatile("s_wait_asynccnt 0x4" ::: "memory");
    } else {
      asm volatile("s_wait_asynccnt 0x0" ::: "memory");
    }
    __syncthreads();  // whole block sees buffer bt filled

    // S = Q * K^T  (two 16x16 tiles covering 32 keys)
    v8f s0 = vzero(), s1 = vzero();
#pragma unroll
    for (int ks = 0; ks < 4; ++ks) {
      v16bf b0 = load_bt_bf16(&Kt[bt][0        + ks * 32], 136, lane);
      v16bf b1 = load_bt_bf16(&Kt[bt][16 * 136 + ks * 32], 136, lane);
      s0 = wmma_bf16(aq[ks], b0, s0);
      s1 = wmma_bf16(aq[ks], b1, s1);
    }

    // online softmax: rows live across a 16-lane half, per-lane 8 rows
    float alpha[8];
#pragma unroll
    for (int r = 0; r < 8; ++r) {
      const float s0r = s0[r] * scale;
      const float s1r = s1[r] * scale;
      float m = fmaxf(s0r, s1r);
#pragma unroll
      for (int off = 1; off < 16; off <<= 1) m = fmaxf(m, __shfl_xor(m, off, 16));
      const float Mn = fmaxf(M[r], m);
      alpha[r] = __expf(M[r] - Mn);
      M[r] = Mn;
      const float p0 = __expf(s0r - Mn);
      const float p1 = __expf(s1r - Mn);
      float rs = p0 + p1;
#pragma unroll
      for (int off = 1; off < 16; off <<= 1) rs += __shfl_xor(rs, off, 16);
      L[r] = L[r] * alpha[r] + rs;
      // stash P in D-layout for the LDS transpose
      Pw[(r + half * 8) * 40 + col]      = (__bf16)p0;
      Pw[(r + half * 8) * 40 + 16 + col] = (__bf16)p1;
    }
#pragma unroll
    for (int t = 0; t < 8; ++t)
#pragma unroll
      for (int r = 0; r < 8; ++r) O[t][r] *= alpha[r];

    // wave-local LDS RAW: drain DS stores before re-reading in A layout
    asm volatile("s_wait_dscnt 0" ::: "memory");
    v16bf pa = load_a_bf16(Pw, 40, lane);

    // O += P * V  (8 channel tiles)
#pragma unroll
    for (int t = 0; t < 8; ++t) {
      v16bf bv = load_bt_bf16(&Vt[bt][(t * 16) * 40], 40, lane);
      O[t] = wmma_bf16(pa, bv, O[t]);
    }
  }

  // epilogue: normalize and write [n][c] bf16
  float invL[8];
#pragma unroll
  for (int r = 0; r < 8; ++r) invL[r] = 1.f / L[r];
  __bf16* hb = ho + (size_t)b * NTOK * CCH;
#pragma unroll
  for (int t = 0; t < 8; ++t)
#pragma unroll
    for (int r = 0; r < 8; ++r)
      hb[(size_t)(q0 + r + half * 8) * CCH + t * 16 + col] = (__bf16)(O[t][r] * invL[r]);
}

// ---------------- Kernel 4: proj GEMM + bias + residual ----------------
__global__ __launch_bounds__(256) void proj_k(const float* __restrict__ x,
                                              const __bf16* __restrict__ hin,
                                              const float* __restrict__ wproj,
                                              const float* __restrict__ bproj,
                                              float* __restrict__ out) {
  const int lane = threadIdx.x & 31;
  const int widx = threadIdx.x >> 5;
  const int wg   = blockIdx.x * 8 + widx;
  const int b    = wg / (NTOK / 16);
  const int tb   = (wg % (NTOK / 16)) * 16;
  const int half = lane >> 4;
  const int col  = lane & 15;

  const __bf16* hb = hin + (size_t)b * NTOK * CCH;
  v16bf a[4];
#pragma unroll
  for (int kt = 0; kt < 4; ++kt)
    a[kt] = load_a_bf16(hb + (size_t)tb * CCH + kt * 32, CCH, lane);

  const float* xb = x   + (size_t)b * CCH * NTOK;
  float*       ob = out + (size_t)b * CCH * NTOK;

#pragma unroll
  for (int ot = 0; ot < 8; ++ot) {
    v8f acc = vzero();
#pragma unroll
    for (int kt = 0; kt < 4; ++kt) {
      v16bf bt = load_bt_f32(wproj + (size_t)(ot * 16) * CCH + kt * 32, CCH, lane);
      acc = wmma_bf16(a[kt], bt, acc);
    }
    const int o = ot * 16 + col;
    const float bias = bproj[o];
    const size_t basei = (size_t)o * NTOK + tb + half * 8;
#pragma unroll
    for (int r = 0; r < 8; ++r)
      ob[basei + r] = xb[basei + r] + acc[r] + bias;
  }
}

extern "C" void kernel_launch(void* const* d_in, const int* in_sizes, int n_in,
                              void* d_out, int out_size, void* d_ws, size_t ws_size,
                              hipStream_t stream) {
  (void)in_sizes; (void)n_in; (void)out_size; (void)ws_size;
  const float* x     = (const float*)d_in[0];
  const float* gamma = (const float*)d_in[1];
  const float* beta  = (const float*)d_in[2];
  const float* wqkv  = (const float*)d_in[3];
  const float* bqkv  = (const float*)d_in[4];
  const float* wproj = (const float*)d_in[5];
  const float* bproj = (const float*)d_in[6];
  float* out = (float*)d_out;

  char* w = (char*)d_ws;
  float*  stats = (float*)w;                            // 32 floats
  __bf16* qws   = (__bf16*)(w + 256);                   // [B][N][C]
  __bf16* kws   = qws  + (size_t)BB * NTOK * CCH;       // [B][N][C]
  __bf16* vtws  = kws  + (size_t)BB * NTOK * CCH;       // [B][C][N]
  __bf16* hws   = vtws + (size_t)BB * NTOK * CCH;       // [B][N][C]

  gn_stats_k<<<BB * GRP, 256, 0, stream>>>(x, stats);
  qkv_k<<<128, 256, 0, stream>>>(x, gamma, beta, wqkv, bqkv, stats, qws, kws, vtws);
  attn_k<<<128, 256, 0, stream>>>(qws, kws, vtws, hws);
  proj_k<<<128, 256, 0, stream>>>(x, hws, wproj, bproj, out);
}
